// Decoder_76862734729499
// MI455X (gfx1250) — compile-verified, matrix operand order checked by reading, not run on hardware
//
#include <hip/hip_runtime.h>

typedef __attribute__((ext_vector_type(16))) __bf16 v16bf;
typedef __attribute__((ext_vector_type(8)))  __bf16 v8bf;
typedef __attribute__((ext_vector_type(8)))  float  v8f;

static constexpr int E_ = 64, A_ = 8, L_ = 2048, D_ = 128, H_ = 8;
static constexpr int TILE = 128;
static constexpr int NTILE = L_ / TILE;     // 16
static constexpr int ROWB = 272;            // 136 bf16 per row (16B aligned, staggered)

// ws layout (bytes)
static constexpr size_t WS_WKC   = 0;        // f32 [129][128]
static constexpr size_t WS_WVC   = 66048;    // f32 [129][128]
static constexpr size_t WS_WQC   = 132096;   // f32 [386][128]
static constexpr size_t WS_WVO   = 329728;   // f32 [8][128][128]
static constexpr size_t WS_BIASO = 854016;   // f32 [128]

union U16 { unsigned u[8]; v16bf v; };

__device__ __forceinline__ unsigned short f2bf(float f) {
  __bf16 x = (__bf16)f;
  return __builtin_bit_cast(unsigned short, x);
}
__device__ __forceinline__ unsigned pack2bf(float a, float b) {
  return (unsigned)f2bf(a) | ((unsigned)f2bf(b) << 16);
}

// Wcomb[i][n] = sum_d Wproj[i][d] * Whead[n/16][d][n%16]; i in [0, rows)
__global__ void combine_kernel(const float* __restrict__ Wproj,
                               const float* __restrict__ Whead,
                               float* __restrict__ outc, int rows) {
  int idx = blockIdx.x * blockDim.x + threadIdx.x;
  if (idx >= rows * 128) return;
  int i = idx >> 7, n = idx & 127;
  int h = n >> 4, j = n & 15;
  float acc = 0.f;
  #pragma unroll 8
  for (int d = 0; d < 128; ++d)
    acc += Wproj[i * 128 + d] * Whead[h * 2048 + d * 16 + j];
  outc[i * 128 + n] = acc;
}

// Wvo[h][k][d] = sum_j WvComb[k][h*16+j] * Wo[(h*16+j)][d];  bias_out[d] = sum_n WvComb[128][n]*Wo[n][d]
__global__ void wvo_kernel(const float* __restrict__ WvComb,
                           const float* __restrict__ Wo,
                           float* __restrict__ Wvo,
                           float* __restrict__ biasO) {
  int idx = blockIdx.x * blockDim.x + threadIdx.x;
  if (idx < 8 * 128 * 128) {
    int h = idx >> 14, k = (idx >> 7) & 127, d = idx & 127;
    float acc = 0.f;
    #pragma unroll
    for (int j = 0; j < 16; ++j)
      acc += WvComb[k * 128 + h * 16 + j] * Wo[(h * 16 + j) * 128 + d];
    Wvo[idx] = acc;
  }
  if (idx < 128) {
    float acc = 0.f;
    for (int n = 0; n < 128; ++n)
      acc += WvComb[128 * 128 + n] * Wo[n * 128 + idx];
    biasO[idx] = acc;
  }
}

struct __align__(16) SMem {
  union {
    char  tile[128 * ROWB];       // bf16 E-tile, row-major
    float redAcc[8][8][128];      // [wave][head][k]  (aliased: used after tile loop)
  } big;
  char  uT[16 * ROWB];            // bf16 u^T [head(16, 8 real)][d(128)]
  float qh[128];
  float sb[16];
  float redM[64];                 // [wave][head]
  float redS[64];
  float ctxe[1024];               // [head][k]
  float psum[2][128];
};

__global__ __launch_bounds__(256, 1)
void decoder_attn_kernel(const float* __restrict__ gctx,
                         const float* __restrict__ depot,
                         const float* __restrict__ tbd,
                         const float* __restrict__ loadv,
                         const float* __restrict__ emb,
                         const int*   __restrict__ lens,
                         const float* __restrict__ Wqcomb,
                         const float* __restrict__ WkComb,
                         const float* __restrict__ Wvo,
                         const float* __restrict__ biasO,
                         float* __restrict__ out) {
  __shared__ SMem sm;
  const int tid  = threadIdx.x;
  const int lane = tid & 31;
  const int wv   = tid >> 5;
  const int lrow = lane & 15;
  const int halfSel = lane >> 4;
  const int b = blockIdx.x;
  const int e = b >> 3, a = b & 7;
  const float aidf = (float)a;

  // ---- query head vector (scaled by 1/sqrt(DK)) ----
  if (tid < 128) {
    float acc = 0.f;
    for (int i = 0; i < 128; ++i) acc += gctx [e * 128 + i] * Wqcomb[(i      ) * 128 + tid];
    for (int i = 0; i < 128; ++i) acc += depot[e * 128 + i] * Wqcomb[(128 + i) * 128 + tid];
    for (int i = 0; i < 128; ++i) acc += tbd  [e * 128 + i] * Wqcomb[(256 + i) * 128 + tid];
    acc += loadv[e * 8 + a] * Wqcomb[384 * 128 + tid];
    acc += aidf             * Wqcomb[385 * 128 + tid];
    sm.qh[tid] = acc * 0.25f;
  }
  __syncthreads();

  // ---- u^T[h][d] = sum_j WkComb[d][h*16+j] * qh[h*16+j]  (bf16; heads 8..15 zero) ----
  for (int idx = tid; idx < 2048; idx += 256) {
    int n = idx >> 7, k = idx & 127;
    float val = 0.f;
    if (n < 8) {
      #pragma unroll
      for (int j = 0; j < 16; ++j)
        val += WkComb[k * 128 + n * 16 + j] * sm.qh[n * 16 + j];
    }
    *(unsigned short*)(sm.uT + n * ROWB + k * 2) = f2bf(val);
  }
  // score bias per head (agent-id feature row of WkComb)
  if (tid < 8) {
    float acc = 0.f;
    #pragma unroll
    for (int j = 0; j < 16; ++j)
      acc += WkComb[128 * 128 + tid * 16 + j] * sm.qh[tid * 16 + j];
    sm.sb[tid] = aidf * acc;
  }
  if (tid >= 8 && tid < 16) sm.sb[tid] = 0.f;
  __syncthreads();

  // ---- preload score-B fragments (tile-invariant) ----
  v16bf ufrag[4];
  #pragma unroll
  for (int kk = 0; kk < 4; ++kk) {
    const char* pb = sm.uT + lrow * ROWB + (kk * 32 + halfSel * 8) * 2;
    v8bf lo = *(const v8bf*)pb;
    v8bf hi = *(const v8bf*)(pb + 32);
    ufrag[kk] = __builtin_shufflevector(lo, hi, 0,1,2,3,4,5,6,7,8,9,10,11,12,13,14,15);
  }
  const float sbl = sm.sb[lrow];

  v8f acc[8];
  #pragma unroll
  for (int kg = 0; kg < 8; ++kg)
    #pragma unroll
    for (int i = 0; i < 8; ++i) acc[kg][i] = 0.f;

  float Mh = -1e30f, Sh = 0.f;
  const int len = lens[e * 8 + a];
  // tiles entirely beyond len contribute nothing -> bound the stream by len
  const int ntile_act = (len + TILE - 1) / TILE;   // 1..16
  const float4* embb = (const float4*)(emb + (size_t)b * (size_t)(L_ * D_));

  for (int t = 0; t < ntile_act; ++t) {
    __syncthreads();
    // cooperative load: f32 tile -> bf16 row-major LDS (8x b128 stores per thread)
    const float4* src = embb + (size_t)t * 4096;
    for (int o = tid; o < 2048; o += 256) {
      int r = o >> 4, c8 = o & 15;
      float4 v0 = src[r * 32 + c8 * 2];
      float4 v1 = src[r * 32 + c8 * 2 + 1];
      uint4 pk;
      pk.x = pack2bf(v0.x, v0.y);
      pk.y = pack2bf(v0.z, v0.w);
      pk.z = pack2bf(v1.x, v1.y);
      pk.w = pack2bf(v1.z, v1.w);
      *(uint4*)(sm.big.tile + r * ROWB + c8 * 16) = pk;
    }
    if (t + 1 < ntile_act) {
      const char* nxt = (const char*)(embb + (size_t)(t + 1) * 4096);
      __builtin_prefetch(nxt + tid * 128, 0, 3);
      __builtin_prefetch(nxt + 32768 + tid * 128, 0, 3);
    }
    __syncthreads();

    // wave-uniform skip, forced scalar: s_cmp/s_cbranch, EXEC untouched (WMMA-safe)
    const int wbase = __builtin_amdgcn_readfirstlane(t * TILE + wv * 16);
    if (wbase < len) {
      // A fragments: this wave's 16 rows of E
      v16bf af[4];
      #pragma unroll
      for (int kk = 0; kk < 4; ++kk) {
        const char* pa = sm.big.tile + (wv * 16 + lrow) * ROWB + (kk * 32 + halfSel * 8) * 2;
        v8bf lo = *(const v8bf*)pa;
        v8bf hi = *(const v8bf*)(pa + 32);
        af[kk] = __builtin_shufflevector(lo, hi, 0,1,2,3,4,5,6,7,8,9,10,11,12,13,14,15);
      }

      // scores[m][h] = E @ u : lane = head, VGPR r = row (r / r+8 per half)
      v8f cs;
      #pragma unroll
      for (int i = 0; i < 8; ++i) cs[i] = sbl;
      #pragma unroll
      for (int kk = 0; kk < 4; ++kk)
        cs = __builtin_amdgcn_wmma_f32_16x16x32_bf16(false, af[kk], false, ufrag[kk],
                                                     (short)0, cs, false, false);

      // per-lane online softmax (head = lane, rows in regs)
      float s[8];
      const int nvalid = len - (wbase + halfSel * 8);   // rows r < nvalid are valid
      #pragma unroll
      for (int r = 0; r < 8; ++r) s[r] = (r < nvalid) ? cs[r] : -1e30f;
      float tm = s[0];
      #pragma unroll
      for (int r = 1; r < 8; ++r) tm = fmaxf(tm, s[r]);
      tm = fmaxf(tm, __shfl_xor(tm, 16));   // per-head max over all 16 rows

      float nM = fmaxf(Mh, tm);
      float sc = __expf(Mh - nM);
      float p[8], rs = 0.f;
      #pragma unroll
      for (int r = 0; r < 8; ++r) { p[r] = __expf(s[r] - nM); rs += p[r]; }
      rs += __shfl_xor(rs, 16);
      Sh = Sh * sc + rs;
      Mh = nM;

      // P as A-fragment: already in the right lanes; zero pad heads >= 8 and K=16..31
      const bool live = (lrow < 8);
      U16 pf;
      #pragma unroll
      for (int i = 0; i < 4; ++i) {
        float p0 = live ? p[2 * i]     : 0.f;
        float p1 = live ? p[2 * i + 1] : 0.f;
        pf.u[i] = pack2bf(p0, p1);
        pf.u[i + 4] = 0u;
      }
      // broadcast per-head rescale factors to all lanes
      float scb[8];
      #pragma unroll
      for (int r = 0; r < 8; ++r) scb[r] = __shfl(sc, r);

      const int mbase = wv * 16 + halfSel * 8;
      #pragma unroll
      for (int kg = 0; kg < 8; ++kg) {
        // B fragment: E^T columns (per-lane strided u16 loads, K 16..31 duplicated)
        const char* pc = sm.big.tile + mbase * ROWB + (kg * 16 + lrow) * 2;
        U16 bf_;
        #pragma unroll
        for (int i = 0; i < 4; ++i) {
          unsigned short e0 = *(const unsigned short*)(pc + (2 * i)     * ROWB);
          unsigned short e1 = *(const unsigned short*)(pc + (2 * i + 1) * ROWB);
          bf_.u[i] = (unsigned)e0 | ((unsigned)e1 << 16);
          bf_.u[i + 4] = bf_.u[i];
        }
        v8f c = acc[kg];
        #pragma unroll
        for (int r = 0; r < 8; ++r) c[r] *= scb[r];
        acc[kg] = __builtin_amdgcn_wmma_f32_16x16x32_bf16(false, pf.v, false, bf_.v,
                                                          (short)0, c, false, false);
      }
    }
  }

  // ---- merge waves (redAcc aliases tile region) ----
  __syncthreads();
  if (lane < 16) {
    #pragma unroll
    for (int kg = 0; kg < 8; ++kg)
      #pragma unroll
      for (int r = 0; r < 8; ++r)
        sm.big.redAcc[wv][r][kg * 16 + lrow] = acc[kg][r];
  }
  if (lane < 8) { sm.redM[wv * 8 + lane] = Mh; sm.redS[wv * 8 + lane] = Sh; }
  __syncthreads();

  #pragma unroll
  for (int ii = 0; ii < 4; ++ii) {
    int idx = tid + ii * 256;          // [head][k]
    int h = idx >> 7, k = idx & 127;
    float gM = -1e30f;
    for (int w = 0; w < 8; ++w) gM = fmaxf(gM, sm.redM[w * 8 + h]);
    float gS = 0.f, gC = 0.f;
    for (int w = 0; w < 8; ++w) {
      float sw = sm.redS[w * 8 + h];
      if (sw > 0.f) {
        float f = __expf(sm.redM[w * 8 + h] - gM);
        gS += sw * f;
        gC += sm.big.redAcc[w][h][k] * f;
      }
    }
    sm.ctxe[idx] = (gS > 0.f) ? (gC / gS) : 0.f;
  }
  __syncthreads();

  // ---- final projection through Wvo (8x128 -> 128), split over 2 half-sums ----
  {
    int d = tid & 127, hb = tid >> 7;
    float accO = 0.f;
    for (int j = hb * 512; j < hb * 512 + 512; ++j)
      accO += sm.ctxe[j] * Wvo[j * 128 + d];
    sm.psum[hb][d] = accO;
  }
  __syncthreads();
  if (tid < 128)
    out[(size_t)b * 128 + tid] = sm.psum[0][tid] + sm.psum[1][tid] + aidf * biasO[tid];
}

extern "C" void kernel_launch(void* const* d_in, const int* in_sizes, int n_in,
                              void* d_out, int out_size, void* d_ws, size_t ws_size,
                              hipStream_t stream) {
  (void)in_sizes; (void)n_in; (void)out_size; (void)ws_size;
  const float* gctx    = (const float*)d_in[0];
  const float* depot   = (const float*)d_in[1];
  const float* tbd     = (const float*)d_in[2];
  const float* loadv   = (const float*)d_in[3];
  const float* emb     = (const float*)d_in[4];
  const int*   lens    = (const int*)  d_in[5];
  const float* Wq_proj = (const float*)d_in[6];
  const float* Wk_proj = (const float*)d_in[7];
  const float* Wv_proj = (const float*)d_in[8];
  const float* Wq      = (const float*)d_in[9];
  const float* Wk      = (const float*)d_in[10];
  const float* Wv      = (const float*)d_in[11];
  const float* Wo      = (const float*)d_in[12];

  char* ws = (char*)d_ws;
  float* WkComb = (float*)(ws + WS_WKC);
  float* WvComb = (float*)(ws + WS_WVC);
  float* Wqcomb = (float*)(ws + WS_WQC);
  float* Wvo    = (float*)(ws + WS_WVO);
  float* biasO  = (float*)(ws + WS_BIASO);

  combine_kernel<<<(129 * 128 + 255) / 256, 256, 0, stream>>>(Wk_proj, Wk, WkComb, 129);
  combine_kernel<<<(129 * 128 + 255) / 256, 256, 0, stream>>>(Wv_proj, Wv, WvComb, 129);
  combine_kernel<<<(386 * 128 + 255) / 256, 256, 0, stream>>>(Wq_proj, Wq, Wqcomb, 386);
  wvo_kernel<<<(8 * 128 * 128 + 255) / 256, 256, 0, stream>>>(WvComb, Wo, Wvo, biasO);

  decoder_attn_kernel<<<E_ * A_, 256, 0, stream>>>(
      gctx, depot, tbd, loadv, emb, lens, Wqcomb, WkComb, Wvo, biasO,
      (float*)d_out);
}
